// sLSTMCell_29145648071332
// MI455X (gfx1250) — matrix operand order repaired and checked by reading, not compiled, not run on hardware
//
#include <hip/hip_runtime.h>
#include <hip/hip_bf16.h>

// sLSTM cell for MI455X (gfx1250): fused bf16-WMMA GEMM + stabilized LSTM epilogue.
// B=16384, I=H=512. gates = [X | h_prev] @ [Wxh | Whh]^T + (bxh + bhh)
// Workgroup (256 thr = 8 waves, arranged 2Mx4N): 64 rows x 64 h-cols x 4 gates.
// Each wave: 2 M-tiles x 4 gate tiles = 8 x v_wmma_f32_16x16x32_bf16 per K-step.
// K loop: 32 steps of K=32 (16 over input/Wxh, 16 over h_prev/Whh).
//
// Fragment layouts (ISA 7.12.2):
//   A 16x32 bf16 : lanes<16 -> K{0-7,16-23}, lanes>=16 -> K{8-15,24-31} (8-blocks)
//   B 32x16 bf16 : lane = N col; lanes<16 -> K{0-7,16-23}, lanes>=16 -> K{8-15,24-31} (8-blocks)
//   C/D 16x16 f32: VGPR r -> M=r (lanes<16) / M=r+8 (lanes>=16); N = lane%16

typedef __attribute__((ext_vector_type(16))) __bf16 v16bf;
typedef __attribute__((ext_vector_type(8)))  __bf16 bf16x8;
typedef __attribute__((ext_vector_type(8)))  float  v8f;

#define BDIM    16384
#define HDIM    512
#define TILE_M  64      // batch rows per workgroup
#define TILE_H  64      // h-columns per workgroup (x4 gates = 256 gate cols)
#define TILE_K  32      // K per step
#define LDA     40      // padded LDS row stride (bf16 elems), 80B rows, 16B aligned
#define NKB     32      // 1024 / 32 K-steps total

union FragU { v16bf v; bf16x8 h[2]; };

__global__ __launch_bounds__(256)
void slstm_wmma_kernel(const float* __restrict__ input,
                       const float* __restrict__ h_prev,
                       const float* __restrict__ c_prev,
                       const float* __restrict__ m_prev,
                       const float* __restrict__ n_prev,
                       const float* __restrict__ Wxh,
                       const float* __restrict__ bxh,
                       const float* __restrict__ Whh,
                       const float* __restrict__ bhh,
                       float* __restrict__ out)
{
    __shared__ __attribute__((aligned(32))) __bf16 sA[TILE_M * LDA];          // 64x32 A tile
    __shared__ __attribute__((aligned(32))) __bf16 sW[4 * TILE_H * LDA];      // 256x32 weight tile

    const int tid  = threadIdx.x;
    const int wave = tid >> 5;
    const int lane = tid & 31;

    const int m0 = blockIdx.x * TILE_M;     // batch-row block
    const int h0 = blockIdx.y * TILE_H;     // h-column block

    const int waveM = wave & 1;             // 0..1  (M direction)
    const int waveN = wave >> 1;            // 0..3  (N direction: 16 h-cols each)

    // ---- cooperative-load indices (A: 64x32, W: 256x32 fp32 -> bf16) ----
    const int lrow = tid >> 2;              // 0..63
    const int lkq  = (tid & 3) * 8;         // 0,8,16,24 (8 floats each)

    // ---- per-wave fragment addressing ----
    const int kbaseA = (lane < 16) ? 0 : 8;   // A: 8-element block interleave
    const int kbaseB = (lane < 16) ? 0 : 8;   // B: 8-element block interleave
    const int lm     = lane & 15;

    v8f acc[2][4];
    {
        v8f z = {};
        #pragma unroll
        for (int mt = 0; mt < 2; ++mt)
            #pragma unroll
            for (int g = 0; g < 4; ++g)
                acc[mt][g] = z;
    }

    for (int kb = 0; kb < NKB; ++kb) {
        const float* Asrc = (kb < 16) ? input : h_prev;
        const float* Wsrc = (kb < 16) ? Wxh : Whh;
        const int k0 = (kb & 15) * TILE_K;

        // ---- stage global fp32 -> registers ----
        float4 af[2];
        {
            const float4* g = reinterpret_cast<const float4*>(
                Asrc + (size_t)(m0 + lrow) * HDIM + k0 + lkq);
            af[0] = g[0]; af[1] = g[1];
        }
        float4 wf[4][2];
        #pragma unroll
        for (int p = 0; p < 4; ++p) {
            const int wr   = p * 64 + lrow;                     // LDS weight row 0..255
            const int grow = (wr >> 6) * HDIM + h0 + (wr & 63); // gate*512 + h0 + hc
            const float4* g = reinterpret_cast<const float4*>(
                Wsrc + (size_t)grow * HDIM + k0 + lkq);
            wf[p][0] = g[0]; wf[p][1] = g[1];
        }

        // prefetch next K-slice (global_prefetch_b8)
        if (kb + 1 < NKB) {
            const float* An = (kb + 1 < 16) ? input : h_prev;
            const float* Wn = (kb + 1 < 16) ? Wxh : Whh;
            const int k0n = ((kb + 1) & 15) * TILE_K;
            __builtin_prefetch(An + (size_t)(m0 + lrow) * HDIM + k0n + lkq, 0, 0);
            __builtin_prefetch(Wn + (size_t)(h0 + lrow) * HDIM + k0n + lkq, 0, 0);
        }

        __syncthreads();   // previous iteration's LDS reads complete

        // ---- fp32 -> bf16, store tiles to LDS ----
        {
            const float* f = reinterpret_cast<const float*>(af);
            bf16x8 v;
            #pragma unroll
            for (int j = 0; j < 8; ++j) v[j] = (__bf16)f[j];
            *reinterpret_cast<bf16x8*>(&sA[lrow * LDA + lkq]) = v;
        }
        #pragma unroll
        for (int p = 0; p < 4; ++p) {
            const float* f = reinterpret_cast<const float*>(wf[p]);
            bf16x8 v;
            #pragma unroll
            for (int j = 0; j < 8; ++j) v[j] = (__bf16)f[j];
            *reinterpret_cast<bf16x8*>(&sW[(p * 64 + lrow) * LDA + lkq]) = v;
        }

        __syncthreads();   // tiles visible to all waves

        // ---- A fragments: 2 M-tiles of 16x32 bf16 ----
        FragU a[2];
        #pragma unroll
        for (int mt = 0; mt < 2; ++mt) {
            const int rowA = (waveM * 2 + mt) * 16 + lm;
            a[mt].h[0] = *reinterpret_cast<const bf16x8*>(&sA[rowA * LDA + kbaseA]);
            a[mt].h[1] = *reinterpret_cast<const bf16x8*>(&sA[rowA * LDA + kbaseA + 16]);
        }

        // ---- B fragments per gate, each reused across both M-tiles ----
        #pragma unroll
        for (int g = 0; g < 4; ++g) {
            FragU b;
            const int rowB = g * TILE_H + waveN * 16 + lm;   // column N of B = weight row
            b.h[0] = *reinterpret_cast<const bf16x8*>(&sW[rowB * LDA + kbaseB]);
            b.h[1] = *reinterpret_cast<const bf16x8*>(&sW[rowB * LDA + kbaseB + 16]);
            #pragma unroll
            for (int mt = 0; mt < 2; ++mt) {
                acc[mt][g] = __builtin_amdgcn_wmma_f32_16x16x32_bf16(
                    false, a[mt].v, false, b.v, (short)0, acc[mt][g], false, false);
            }
        }
    }

    // -------- epilogue: stabilized sLSTM update --------
    const int hcol = h0 + waveN * 16 + lm;

    const float bi = bxh[0 * HDIM + hcol] + bhh[0 * HDIM + hcol];
    const float bf = bxh[1 * HDIM + hcol] + bhh[1 * HDIM + hcol];
    const float bz = bxh[2 * HDIM + hcol] + bhh[2 * HDIM + hcol];
    const float bo = bxh[3 * HDIM + hcol] + bhh[3 * HDIM + hcol];

    const size_t plane = (size_t)BDIM * HDIM;
    float* out_h = out;
    float* out_c = out + plane;
    float* out_m = out + 2 * plane;
    float* out_n = out + 3 * plane;

    #pragma unroll
    for (int mt = 0; mt < 2; ++mt) {
        const int mbase = m0 + (waveM * 2 + mt) * 16 + ((lane >> 4) << 3);
        #pragma unroll
        for (int r = 0; r < 8; ++r) {
            const int m = mbase + r;
            const size_t idx = (size_t)m * HDIM + hcol;

            const float ig = acc[mt][0][r] + bi;
            const float fg = acc[mt][1][r] + bf;
            const float zg = acc[mt][2][r] + bz;
            const float og = acc[mt][3][r] + bo;

            const float cp = c_prev[idx];
            const float mp = m_prev[idx];
            const float np = n_prev[idx];

            // z_t = tanh(zg) via exp (saturates correctly at +/-1)
            const float e2z = __expf(2.0f * zg);
            const float z_t = 1.0f - 2.0f / (e2z + 1.0f);
            // o_t = sigmoid(og)
            const float o_t = 1.0f / (1.0f + __expf(-og));
            // log-space stabilizer
            const float m_t = fmaxf(fg + mp, ig);
            const float i_t = __expf(ig - m_t);
            const float f_t = __expf(fg + mp - m_t);

            const float c_t = f_t * cp + i_t * z_t;
            const float n_t = f_t * np + i_t;
            const float h_t = o_t * (c_t / n_t);

            out_h[idx] = h_t;
            out_c[idx] = c_t;
            out_m[idx] = m_t;
            out_n[idx] = n_t;
        }
    }
}

extern "C" void kernel_launch(void* const* d_in, const int* in_sizes, int n_in,
                              void* d_out, int out_size, void* d_ws, size_t ws_size,
                              hipStream_t stream) {
    (void)in_sizes; (void)n_in; (void)out_size; (void)d_ws; (void)ws_size;

    const float* input  = (const float*)d_in[0];
    const float* h_prev = (const float*)d_in[1];
    const float* c_prev = (const float*)d_in[2];
    const float* m_prev = (const float*)d_in[3];
    const float* n_prev = (const float*)d_in[4];
    const float* Wxh    = (const float*)d_in[5];
    const float* bxh    = (const float*)d_in[6];
    const float* Whh    = (const float*)d_in[7];
    const float* bhh    = (const float*)d_in[8];
    float* out = (float*)d_out;

    dim3 grid(BDIM / TILE_M, HDIM / TILE_H);   // (256, 8)
    dim3 block(256);
    slstm_wmma_kernel<<<grid, block, 0, stream>>>(
        input, h_prev, c_prev, m_prev, n_prev, Wxh, bxh, Whh, bhh, out);
}
